// NDGroupedQueryAttention_53858889892403
// MI455X (gfx1250) — matrix operand order repaired and checked by reading, not compiled, not run on hardware
//
#include <hip/hip_runtime.h>
#include <hip/hip_bf16.h>

// ---------------------------------------------------------------------------
// NDGroupedQueryAttention for MI455X (gfx1250), wave32 + WMMA bf16 + async TDM.
// prep weights (group-summed Wq, weights pre-transposed [N][K] bf16)
//  -> 3 projection GEMMs (double-buffered TDM B-staging)
//  -> RoPE/convert (V written pre-transposed [bh][d][key])
//  -> online-softmax attention (double-buffered TDM K/V staging)
//  -> output GEMM.
// Fragment loads are batched ahead of WMMA clusters so LDS waits amortize.
// ---------------------------------------------------------------------------

typedef __attribute__((ext_vector_type(16))) __bf16 bf16x16;
typedef __attribute__((ext_vector_type(8)))  __bf16 bf16x8;
typedef __attribute__((ext_vector_type(8)))  float  f32x8;
typedef __attribute__((ext_vector_type(4)))  float  f32x4;
typedef __attribute__((ext_vector_type(4)))  unsigned int u32x4;
typedef __attribute__((ext_vector_type(8)))  int    i32x8;
typedef __attribute__((ext_vector_type(4)))  int    i32x4;

#define B_      4
#define SQ_     2048
#define SKV_    2048
#define IN_     1024
#define KVH_    4
#define HD_     64
#define DM_     256          // KVH_*HD_

__device__ __forceinline__ bf16x8 ld8(const __bf16* p) {
    return *(const bf16x8*)p;
}

__device__ __forceinline__ bf16x16 mk16(bf16x8 lo, bf16x8 hi) {
    bf16x16 r;
#pragma unroll
    for (int i = 0; i < 8; ++i) { r[i] = lo[i]; r[i + 8] = hi[i]; }
    return r;
}

// ---------------------------------------------------------------------------
// Tensor Data Mover: 2D tile (bf16) global -> LDS (async, TENSORcnt).
// D# packed per CDNA5 ISA 8.3-8.5 (count=1, type=2 image, data_size=2B).
// ---------------------------------------------------------------------------
__device__ __forceinline__ void tdm_load_2d_bf16(unsigned long long gaddr,
                                                 unsigned int lds_addr,
                                                 unsigned int tile_d0,
                                                 unsigned int tile_d1,
                                                 unsigned long long stride_d0)
{
    u32x4 g0;
    g0[0] = 1u;                                        // count=1, user mode
    g0[1] = lds_addr;                                  // lds_addr[31:0]
    g0[2] = (unsigned int)(gaddr & 0xffffffffull);     // global_addr[31:0]
    g0[3] = (unsigned int)((gaddr >> 32) & 0x1ffffffull) | (2u << 30); // addr[56:32] | type=2

    unsigned int td0 = tile_d0;                        // tensor_dim0 (OOB bound)
    unsigned int td1 = tile_d1;                        // tensor_dim1
    i32x8 g1;
    g1[0] = (int)(1u << 16);                           // wg_mask=0, data_size=1 (2 bytes)
    g1[1] = (int)((td0 & 0xffffu) << 16);              // tensor_dim0[15:0] @ bits 63:48
    g1[2] = (int)(((td0 >> 16) & 0xffffu) | ((td1 & 0xffffu) << 16));
    g1[3] = (int)(((td1 >> 16) & 0xffffu) | ((tile_d0 & 0xffffu) << 16)); // tile_dim0
    g1[4] = (int)(tile_d1 & 0xffffu);                  // tile_dim1, tile_dim2=0 (2D)
    g1[5] = (int)(stride_d0 & 0xffffffffull);          // tensor_dim0_stride[31:0]
    g1[6] = (int)((stride_d0 >> 32) & 0xffffull);      // stride[47:32]; dim1_stride lo16 = 0
    g1[7] = 0;

    i32x4 g2; g2[0] = 0; g2[1] = 0; g2[2] = 0; g2[3] = 0;
    i32x4 g3; g3[0] = 0; g3[1] = 0; g3[2] = 0; g3[3] = 0;

#if defined(__clang_major__) && (__clang_major__ >= 23)
    i32x8 g4;
#pragma unroll
    for (int i = 0; i < 8; ++i) g4[i] = 0;
    __builtin_amdgcn_tensor_load_to_lds(g0, g1, g2, g3, g4, 0);
#else
    __builtin_amdgcn_tensor_load_to_lds(g0, g1, g2, g3, 0);
#endif
}

// ---------------------------------------------------------------------------
// Weight prep: Wq summed over groups (RoPE is linear -> group-sum commutes).
// All weights emitted bf16, pre-transposed to [N][K] (TDM-friendly rows).
// ---------------------------------------------------------------------------
__global__ __launch_bounds__(256)
void prep_w_kernel(const float* __restrict__ Wq, const float* __restrict__ Wk,
                   const float* __restrict__ Wv, const float* __restrict__ Wo,
                   __bf16* __restrict__ wqe, __bf16* __restrict__ wkb,
                   __bf16* __restrict__ wvb, __bf16* __restrict__ wob)
{
    int t = blockIdx.x * blockDim.x + threadIdx.x;      // 1024*256 threads
    int k = t & 1023, col = t >> 10;                    // col = h*64+dd (0..255)
    int h = col >> 6, dd = col & 63;
    float s = 0.f;
#pragma unroll
    for (int g = 0; g < 4; ++g)
        s += Wq[((size_t)k * 16 + (h * 4 + g)) * 64 + dd];
    wqe[(size_t)col * IN_ + k] = (__bf16)s;
    wkb[(size_t)col * IN_ + k] = (__bf16)Wk[((size_t)k * 4 + h) * 64 + dd];
    wvb[(size_t)col * IN_ + k] = (__bf16)Wv[((size_t)k * 4 + h) * 64 + dd];
    int k2 = t & 255, n2 = t >> 8;                      // Wo: [256][1024] -> [1024][256]
    wob[(size_t)n2 * DM_ + k2] = (__bf16)Wo[(size_t)k2 * IN_ + n2];
}

// ---------------------------------------------------------------------------
// GEMM:  C[M x N] (f32) = A[M x K] (f32 -> bf16 in LDS) x Bt[N x K] (bf16).
// 128 threads (4 waves), tile 64x128, K-step 32.
// B staged by double-buffered TDM; all 8 B fragments loaded before the
// 8-WMMA burst so one s_wait_dscnt covers the cluster.
// ---------------------------------------------------------------------------
__global__ __launch_bounds__(128)
void gemm_f32xbf16_kernel(const float* __restrict__ A, const __bf16* __restrict__ Bt,
                          float* __restrict__ C, int M, int N, int K)
{
    __shared__ __align__(16) __bf16 As[2][64][32];     // [buf][m][k]
    __shared__ __align__(16) __bf16 Bs[2][128][32];    // [buf][n][k]

    const int t = threadIdx.x;
    const int wave = t >> 5, lane = t & 31;
    const int half = lane >> 4, ln = lane & 15;
    const int mblk = blockIdx.x * 64, nblk = blockIdx.y * 128;
    const unsigned int bs_lds[2] = { (unsigned int)(size_t)&Bs[0][0][0],
                                     (unsigned int)(size_t)&Bs[1][0][0] };
    const __bf16* bbase = Bt + (size_t)nblk * K;

    f32x8 acc[8];
#pragma unroll
    for (int c = 0; c < 8; ++c)
#pragma unroll
        for (int r = 0; r < 8; ++r) acc[c][r] = 0.f;

    const int arow = t >> 1, aseg = (t & 1) * 16;
    const float* abase = A + (size_t)(mblk + arow) * K + aseg;
    const int nk = K / 32;

    // prologue: chunk 0 -> buffer 0
    if (wave == 0)
        tdm_load_2d_bf16((unsigned long long)(size_t)bbase,
                         bs_lds[0], 32u, 128u, (unsigned long long)K);

    for (int i = 0; i < nk; ++i) {
        const int cur = i & 1, k0 = i * 32;
        // async: TDM pulls next B tile into the alternate buffer
        if (wave == 0 && i + 1 < nk)
            tdm_load_2d_bf16((unsigned long long)(size_t)(bbase + k0 + 32),
                             bs_lds[cur ^ 1], 32u, 128u, (unsigned long long)K);
        {   // stage A chunk i: 16 floats/thread, f32->bf16, 2x b128 LDS stores
            const float* src = abase + k0;
            f32x4 f0 = *(const f32x4*)(src);
            f32x4 f1 = *(const f32x4*)(src + 4);
            f32x4 f2 = *(const f32x4*)(src + 8);
            f32x4 f3 = *(const f32x4*)(src + 12);
            bf16x8 v0, v1;
#pragma unroll
            for (int j = 0; j < 4; ++j) {
                v0[j] = (__bf16)f0[j]; v0[j + 4] = (__bf16)f1[j];
                v1[j] = (__bf16)f2[j]; v1[j + 4] = (__bf16)f3[j];
            }
            *(bf16x8*)&As[cur][arow][aseg]     = v0;
            *(bf16x8*)&As[cur][arow][aseg + 8] = v1;
            if (k0 + 32 < K) __builtin_prefetch(abase + k0 + 32, 0, 0);
        }
        // in-order TENSORcnt: <=1 outstanding means chunk i's tile has landed
        if (wave == 0) {
            if (i + 1 < nk) __builtin_amdgcn_s_wait_tensorcnt(1);
            else            __builtin_amdgcn_s_wait_tensorcnt(0);
        }
        __syncthreads();

        // batch all fragment loads, then an 8-WMMA burst
        bf16x16 af = mk16(ld8(&As[cur][wave * 16 + ln][half * 8]),
                          ld8(&As[cur][wave * 16 + ln][half * 8 + 16]));
        bf16x16 bfr[8];
#pragma unroll
        for (int c = 0; c < 8; ++c)
            bfr[c] = mk16(ld8(&Bs[cur][c * 16 + ln][half * 8]),
                          ld8(&Bs[cur][c * 16 + ln][half * 8 + 16]));
#pragma unroll
        for (int c = 0; c < 8; ++c)
            acc[c] = __builtin_amdgcn_wmma_f32_16x16x32_bf16(
                false, af, false, bfr[c], (short)0, acc[c], false, false);
        __syncthreads();   // retire buffer `cur` before iter i+1 overwrites it
    }

#pragma unroll
    for (int c = 0; c < 8; ++c)
#pragma unroll
        for (int r = 0; r < 8; ++r) {
            int row = mblk + wave * 16 + half * 8 + r;
            int col = nblk + c * 16 + ln;
            C[(size_t)row * N + col] = acc[c][r];
        }
}

// ---------------------------------------------------------------------------
// RoPE + scale + bf16 convert.  Thread bits: s fastest (coalesced V writes).
// q,k out: [b][h][s][64].  v out: TRANSPOSED [b][h][d][key].
// ---------------------------------------------------------------------------
__global__ __launch_bounds__(256)
void rope_cvt_kernel(const float* __restrict__ qp, const float* __restrict__ kp,
                     const float* __restrict__ vp,
                     const int* __restrict__ qc, const int* __restrict__ kc,
                     __bf16* __restrict__ qr, __bf16* __restrict__ kr,
                     __bf16* __restrict__ vbt)
{
    int t = blockIdx.x * blockDim.x + threadIdx.x;   // 2^20 threads
    int s = t & 2047;
    int h = (t >> 11) & 3;
    int j = (t >> 13) & 31;
    int b = t >> 18;
    int bh = b * 4 + h;

    int axis = j >> 4, i = j & 15;                   // dpa=32 -> 16 freqs per axis
    float invf = __powf(10000.f, -(float)i / 16.f);

    size_t inb  = ((size_t)(b * 2048 + s)) * 256 + h * 64 + 2 * j;
    size_t outb = (((size_t)bh) * 2048 + s) * 64 + 2 * j;

    {   // q: rotate + pre-scale by 1/sqrt(64)
        float th = (float)qc[(b * 2048 + s) * 2 + axis] * invf;
        float sn, cs; __sincosf(th, &sn, &cs);
        float x1 = qp[inb], x2 = qp[inb + 1];
        qr[outb]     = (__bf16)(0.125f * (x1 * cs - x2 * sn));
        qr[outb + 1] = (__bf16)(0.125f * (x1 * sn + x2 * cs));
    }
    {   // k: rotate
        float th = (float)kc[(b * 2048 + s) * 2 + axis] * invf;
        float sn, cs; __sincosf(th, &sn, &cs);
        float x1 = kp[inb], x2 = kp[inb + 1];
        kr[outb]     = (__bf16)(x1 * cs - x2 * sn);
        kr[outb + 1] = (__bf16)(x1 * sn + x2 * cs);
    }
    vbt[((size_t)bh * 64 + 2 * j)     * 2048 + s] = (__bf16)vp[inb];
    vbt[((size_t)bh * 64 + 2 * j + 1) * 2048 + s] = (__bf16)vp[inb + 1];
}

// ---------------------------------------------------------------------------
// Attention with online softmax.  grid = (SQ/64, B*KVH), 128 threads (4 waves).
// K [32 key][64 d] and V [64 d][32 key] tiles double-buffered via TDM:
// the K/V pair for chunk i+1 is in flight during WMMA+softmax on chunk i.
// ---------------------------------------------------------------------------
__global__ __launch_bounds__(128)
void attn_kernel(const __bf16* __restrict__ Qr, const __bf16* __restrict__ Kr,
                 const __bf16* __restrict__ Vt_g, float* __restrict__ AO)
{
    const int t = threadIdx.x;
    const int wave = t >> 5, lane = t & 31;
    const int half = lane >> 4, ln = lane & 15;
    const int qtile = blockIdx.x, bh = blockIdx.y;
    const int b = bh >> 2, h = bh & 3;

    const __bf16* Qbh = Qr   + (size_t)bh * SKV_ * 64;
    const __bf16* Kbh = Kr   + (size_t)bh * SKV_ * 64;
    const __bf16* Vbh = Vt_g + (size_t)bh * 64 * SKV_;   // [d][key]

    __shared__ __align__(16) __bf16 Ks[2][32][64];       // [buf][key][d]
    __shared__ __align__(16) __bf16 Vs[2][64][32];       // [buf][d][key]
    __shared__ __align__(16) __bf16 Ps[4][16][32];       // per-wave P tile [m][k]
    const unsigned int ks_lds[2] = { (unsigned int)(size_t)&Ks[0][0][0],
                                     (unsigned int)(size_t)&Ks[1][0][0] };
    const unsigned int vs_lds[2] = { (unsigned int)(size_t)&Vs[0][0][0],
                                     (unsigned int)(size_t)&Vs[1][0][0] };

    // Q fragments for this wave's 16 rows (held for the whole key loop)
    const __bf16* qrow = Qbh + (size_t)(qtile * 64 + wave * 16 + ln) * 64;
    bf16x16 qf0 = mk16(ld8(qrow + half * 8),      ld8(qrow + half * 8 + 16));
    bf16x16 qf1 = mk16(ld8(qrow + 32 + half * 8), ld8(qrow + 32 + half * 8 + 16));

    f32x8 o0, o1, o2, o3;
    float mi[8], li[8];
#pragma unroll
    for (int r = 0; r < 8; ++r) {
        o0[r] = 0.f; o1[r] = 0.f; o2[r] = 0.f; o3[r] = 0.f;
        mi[r] = -1e30f; li[r] = 0.f;
    }

    const int nchunk = SKV_ / 32;
    if (wave == 0) {   // prologue: chunk 0 -> buffer 0
        tdm_load_2d_bf16((unsigned long long)(size_t)Kbh, ks_lds[0], 64u, 32u, 64ull);
        tdm_load_2d_bf16((unsigned long long)(size_t)Vbh, vs_lds[0], 32u, 64u,
                         (unsigned long long)SKV_);
    }

    for (int i = 0; i < nchunk; ++i) {
        const int cur = i & 1, c0 = i * 32;
        if (wave == 0) {
            if (i + 1 < nchunk) {
                tdm_load_2d_bf16((unsigned long long)(size_t)(Kbh + (size_t)(c0 + 32) * 64),
                                 ks_lds[cur ^ 1], 64u, 32u, 64ull);
                tdm_load_2d_bf16((unsigned long long)(size_t)(Vbh + c0 + 32),
                                 vs_lds[cur ^ 1], 32u, 64u, (unsigned long long)SKV_);
                __builtin_amdgcn_s_wait_tensorcnt(2);   // pair i landed, pair i+1 in flight
            } else {
                __builtin_amdgcn_s_wait_tensorcnt(0);
            }
        }
        __syncthreads();

        // scores S = Q(16x64) * K_chunk^T(64x32): batch 4 K-frag loads, then
        // a 4-WMMA burst (two N-tiles x two K-steps)
        f32x8 s0, s1;
#pragma unroll
        for (int r = 0; r < 8; ++r) { s0[r] = 0.f; s1[r] = 0.f; }
        {
            bf16x16 kb[4];
            kb[0] = mk16(ld8(&Ks[cur][ln][half * 8]),           ld8(&Ks[cur][ln][half * 8 + 16]));
            kb[1] = mk16(ld8(&Ks[cur][ln][32 + half * 8]),      ld8(&Ks[cur][ln][32 + half * 8 + 16]));
            kb[2] = mk16(ld8(&Ks[cur][16 + ln][half * 8]),      ld8(&Ks[cur][16 + ln][half * 8 + 16]));
            kb[3] = mk16(ld8(&Ks[cur][16 + ln][32 + half * 8]), ld8(&Ks[cur][16 + ln][32 + half * 8 + 16]));
            s0 = __builtin_amdgcn_wmma_f32_16x16x32_bf16(false, qf0, false, kb[0], (short)0, s0, false, false);
            s0 = __builtin_amdgcn_wmma_f32_16x16x32_bf16(false, qf1, false, kb[1], (short)0, s0, false, false);
            s1 = __builtin_amdgcn_wmma_f32_16x16x32_bf16(false, qf0, false, kb[2], (short)0, s1, false, false);
            s1 = __builtin_amdgcn_wmma_f32_16x16x32_bf16(false, qf1, false, kb[3], (short)0, s1, false, false);
        }

        // online softmax update (rows live across one 16-lane half)
#pragma unroll
        for (int r = 0; r < 8; ++r) {
            float v = fmaxf(s0[r], s1[r]);
#pragma unroll
            for (int m = 1; m < 16; m <<= 1) v = fmaxf(v, __shfl_xor(v, m));
            float mnew = fmaxf(mi[r], v);
            float alpha = __expf(mi[r] - mnew);
            float p0 = __expf(s0[r] - mnew);
            float p1 = __expf(s1[r] - mnew);
            float rs = p0 + p1;
#pragma unroll
            for (int m = 1; m < 16; m <<= 1) rs += __shfl_xor(rs, m);
            li[r] = li[r] * alpha + rs;
            mi[r] = mnew;
            o0[r] *= alpha; o1[r] *= alpha; o2[r] *= alpha; o3[r] *= alpha;
            Ps[wave][half * 8 + r][ln]      = (__bf16)p0;
            Ps[wave][half * 8 + r][16 + ln] = (__bf16)p1;
        }
        __syncthreads();

        // P fragment (16x32); batch 4 V-frag loads, then a 4-WMMA burst
        bf16x16 pf = mk16(ld8(&Ps[wave][ln][half * 8]),
                          ld8(&Ps[wave][ln][half * 8 + 16]));
        {
            bf16x16 vf[4];
#pragma unroll
            for (int c = 0; c < 4; ++c)
                vf[c] = mk16(ld8(&Vs[cur][c * 16 + ln][half * 8]),
                             ld8(&Vs[cur][c * 16 + ln][half * 8 + 16]));
            o0 = __builtin_amdgcn_wmma_f32_16x16x32_bf16(false, pf, false, vf[0], (short)0, o0, false, false);
            o1 = __builtin_amdgcn_wmma_f32_16x16x32_bf16(false, pf, false, vf[1], (short)0, o1, false, false);
            o2 = __builtin_amdgcn_wmma_f32_16x16x32_bf16(false, pf, false, vf[2], (short)0, o2, false, false);
            o3 = __builtin_amdgcn_wmma_f32_16x16x32_bf16(false, pf, false, vf[3], (short)0, o3, false, false);
        }
        __syncthreads();   // retire K/V buffer `cur` before iter i+1 overwrites it
    }

    // epilogue: normalize and write f32 [b][s][h*64+dd]
#pragma unroll
    for (int r = 0; r < 8; ++r) {
        int s = qtile * 64 + wave * 16 + half * 8 + r;
        float inv = 1.f / li[r];
        size_t base = ((size_t)(b * 2048 + s)) * 256 + h * 64;
        AO[base + 0 * 16 + ln] = o0[r] * inv;
        AO[base + 1 * 16 + ln] = o1[r] * inv;
        AO[base + 2 * 16 + ln] = o2[r] * inv;
        AO[base + 3 * 16 + ln] = o3[r] * inv;
    }
}

// ---------------------------------------------------------------------------
extern "C" void kernel_launch(void* const* d_in, const int* in_sizes, int n_in,
                              void* d_out, int out_size, void* d_ws, size_t ws_size,
                              hipStream_t stream) {
    const float* q  = (const float*)d_in[0];
    const int*   qc = (const int*)  d_in[1];
    const float* kv = (const float*)d_in[2];
    const int*   kc = (const int*)  d_in[3];
    const float* Wq = (const float*)d_in[4];
    const float* Wk = (const float*)d_in[5];
    const float* Wv = (const float*)d_in[6];
    const float* Wo = (const float*)d_in[7];
    float* out = (float*)d_out;

    // workspace layout
    char* w = (char*)d_ws;
    __bf16* wqe = (__bf16*)w; w += (size_t)IN_ * DM_ * 2;      // [256][1024] bf16
    __bf16* wkb = (__bf16*)w; w += (size_t)IN_ * DM_ * 2;
    __bf16* wvb = (__bf16*)w; w += (size_t)IN_ * DM_ * 2;
    __bf16* wob = (__bf16*)w; w += (size_t)DM_ * IN_ * 2;      // [1024][256] bf16
    float*  qp  = (float*)w;  w += (size_t)B_ * SQ_ * DM_ * 4; // 8 MB (reused as AO)
    float*  kp  = (float*)w;  w += (size_t)B_ * SKV_ * DM_ * 4;
    float*  vp  = (float*)w;  w += (size_t)B_ * SKV_ * DM_ * 4;
    __bf16* qr  = (__bf16*)w; w += (size_t)B_ * SQ_ * DM_ * 2; // [bh][s][64]
    __bf16* kr  = (__bf16*)w; w += (size_t)B_ * SKV_ * DM_ * 2;
    __bf16* vbt = (__bf16*)w; w += (size_t)B_ * SKV_ * DM_ * 2; // [bh][64][2048]
    float*  ao  = qp;  // qp is dead after rope_cvt

    prep_w_kernel<<<1024, 256, 0, stream>>>(Wq, Wk, Wv, Wo, wqe, wkb, wvb, wob);

    // projections: [8192 x 1024] x [1024 x 256] (B pre-transposed)
    gemm_f32xbf16_kernel<<<dim3(B_ * SQ_ / 64, DM_ / 128), 128, 0, stream>>>(
        q, wqe, qp, B_ * SQ_, DM_, IN_);
    gemm_f32xbf16_kernel<<<dim3(B_ * SKV_ / 64, DM_ / 128), 128, 0, stream>>>(
        kv, wkb, kp, B_ * SKV_, DM_, IN_);
    gemm_f32xbf16_kernel<<<dim3(B_ * SKV_ / 64, DM_ / 128), 128, 0, stream>>>(
        kv, wvb, vp, B_ * SKV_, DM_, IN_);

    rope_cvt_kernel<<<4096, 256, 0, stream>>>(qp, kp, vp, qc, kc, qr, kr, vbt);

    attn_kernel<<<dim3(SQ_ / 64, B_ * KVH_), 128, 0, stream>>>(qr, kr, vbt, ao);

    // output projection: [8192 x 256] x [256 x 1024]
    gemm_f32xbf16_kernel<<<dim3(B_ * SQ_ / 64, IN_ / 128), 128, 0, stream>>>(
        ao, wob, out, B_ * SQ_, IN_, DM_);
}
